// VectorQuantizerEMA_446676599465
// MI455X (gfx1250) — matrix operand order repaired and checked by reading, not compiled, not run on hardware
//
#include <hip/hip_runtime.h>
#include <hip/hip_bf16.h>

#define B_ 32
#define S_ 1024
#define D_ 256
#define K_ 1024
#define N_ (B_ * S_)

#define DECAY_ 0.99f
#define EPS_ 1e-05f
#define LN_EPS_ 1e-05f
#define COMMIT_ 0.25f

// ---------------- WMMA types (CDNA5 / gfx1250, wave32) ----------------
typedef __attribute__((ext_vector_type(16))) __bf16 v16bf;
typedef __attribute__((ext_vector_type(8)))  float  v8f;

struct alignas(16) U4 { unsigned int w[4]; };
union BFrag { v16bf v; U4 q[2]; };

static __device__ __forceinline__ unsigned short f2bf(float f) {
    unsigned int u = __float_as_uint(f);
    unsigned int r = u + 0x7FFFu + ((u >> 16) & 1u);   // round-to-nearest-even
    return (unsigned short)(r >> 16);
}

// --- CDNA5 async global->LDS copy (ASYNCcnt), GVS mode: SGPR base + VGPR offset
static __device__ __forceinline__ void async_ld16(unsigned lds_addr, unsigned goff,
                                                  unsigned long long base) {
    asm volatile("global_load_async_to_lds_b128 %0, %1, %2"
                 :: "v"(lds_addr), "v"(goff), "s"(base) : "memory");
}
static __device__ __forceinline__ void wait_async16() {
    asm volatile("s_wait_asynccnt 0x10" ::: "memory");
}
static __device__ __forceinline__ void wait_async0() {
    asm volatile("s_wait_asynccnt 0x0" ::: "memory");
}

// ---------------- 0: zero scratch ----------------
__global__ void vq_zero_kernel(float* __restrict__ p, int n) {
    int i = blockIdx.x * blockDim.x + threadIdx.x;
    if (i < n) p[i] = 0.0f;
}

// ---------------- 1: LayerNorm -> bf16 feats + ||f||^2 ----------------
__global__ __launch_bounds__(256) void vq_ln_kernel(
    const float* __restrict__ x, const float* __restrict__ g, const float* __restrict__ b,
    unsigned short* __restrict__ featsBF, float* __restrict__ f2out)
{
    __shared__ float red[8];
    __shared__ float bc;
    const int n = blockIdx.x, d = threadIdx.x;
    const int wave = d >> 5, lane = d & 31;
    float v = x[(size_t)n * D_ + d];

    float s = v;
    #pragma unroll
    for (int m = 16; m >= 1; m >>= 1) s += __shfl_xor(s, m, 32);
    if (lane == 0) red[wave] = s;
    __syncthreads();
    if (d == 0) { float t = 0.f; for (int i = 0; i < 8; ++i) t += red[i]; bc = t * (1.0f / D_); }
    __syncthreads();
    const float mu = bc;
    __syncthreads();

    float dv = v - mu;
    s = dv * dv;
    #pragma unroll
    for (int m = 16; m >= 1; m >>= 1) s += __shfl_xor(s, m, 32);
    if (lane == 0) red[wave] = s;
    __syncthreads();
    if (d == 0) { float t = 0.f; for (int i = 0; i < 8; ++i) t += red[i]; bc = t * (1.0f / D_); }
    __syncthreads();
    const float rs = rsqrtf(bc + LN_EPS_);
    __syncthreads();

    const float f = dv * rs * g[d] + b[d];
    featsBF[(size_t)n * D_ + d] = f2bf(f);

    s = f * f;
    #pragma unroll
    for (int m = 16; m >= 1; m >>= 1) s += __shfl_xor(s, m, 32);
    if (lane == 0) red[wave] = s;
    __syncthreads();
    if (d == 0) { float t = 0.f; for (int i = 0; i < 8; ++i) t += red[i]; f2out[n] = t; }
}

// ---------------- 2: codebook -> bf16 + ||w||^2 ----------------
__global__ __launch_bounds__(256) void vq_wprep_kernel(
    const float* __restrict__ w, unsigned short* __restrict__ wBF, float* __restrict__ w2out)
{
    __shared__ float red[8];
    const int k = blockIdx.x, d = threadIdx.x;
    const int wave = d >> 5, lane = d & 31;
    float v = w[(size_t)k * D_ + d];
    wBF[(size_t)k * D_ + d] = f2bf(v);
    float s = v * v;
    #pragma unroll
    for (int m = 16; m >= 1; m >>= 1) s += __shfl_xor(s, m, 32);
    if (lane == 0) red[wave] = s;
    __syncthreads();
    if (d == 0) { float t = 0.f; for (int i = 0; i < 8; ++i) t += red[i]; w2out[k] = t; }
}

// ---------------- 3: WMMA distances + argmin over K ----------------
// Block = 256 thr (8 waves) = 64 rows. Codebook streamed through LDS in 8 chunks
// of 128 codes (64 KB), double-buffered via global_load_async_to_lds_b128.
// Wave w: row group (w&3) [16 rows], column half (w>>2) [64 of 128 chunk codes].
#define CHUNK_CODES 128
#define CHUNK_SH    (CHUNK_CODES * D_)          // bf16 elements per chunk
#define CHUNK_BYTES (CHUNK_SH * 2)              // 65536

__global__ __launch_bounds__(256) void vq_dist_argmin_kernel(
    const unsigned short* __restrict__ featsBF, const float* __restrict__ f2,
    const unsigned short* __restrict__ wBF, const float* __restrict__ w2,
    float* __restrict__ minv_out, int* __restrict__ mini_out)
{
    extern __shared__ unsigned short sbuf[];    // 2 * 64 KB, dynamic LDS
    __shared__ float sv[8 * 16];
    __shared__ int   si[8 * 16];

    const int tid  = threadIdx.x;
    const int wave = tid >> 5, lane = tid & 31;
    const int rg   = wave & 3;                  // row group 0..3
    const int ch   = wave >> 2;                 // column half 0..1
    const int m0   = blockIdx.x * 64;
    const int hi   = (lane >> 4) & 1;
    const int lrow = lane & 15;

    const unsigned long long wbase = (unsigned long long)(size_t)wBF;
    const unsigned lbase0 = (unsigned)(size_t)sbuf;   // LDS aperture keeps offset in addr[31:0]

    // prologue: async-load chunk 0 into buffer 0 (16 x b128 per thread)
    #pragma unroll
    for (int i = 0; i < 16; ++i) {
        unsigned u = (unsigned)tid + 256u * (unsigned)i;    // 16-byte unit id
        async_ld16(lbase0 + u * 16u, u * 16u, wbase);
    }

    // A fragments for this wave's 16 rows (ISA 16-bit A 16x32 layout)
    BFrag A[8];
    {
        const unsigned short* arow = featsBF + (size_t)(m0 + rg * 16 + lrow) * D_;
        #pragma unroll
        for (int c = 0; c < 8; ++c) {
            const int base = c * 32 + hi * 8;
            A[c].q[0] = *reinterpret_cast<const U4*>(arow + base);
            A[c].q[1] = *reinterpret_cast<const U4*>(arow + base + 16);
        }
    }

    float minv[8]; int mini[8];
    #pragma unroll
    for (int r = 0; r < 8; ++r) { minv[r] = 3.0e38f; mini[r] = 0; }

    for (int c = 0; c < 8; ++c) {
        if (c < 7) {    // prefetch next chunk into the other buffer
            const unsigned lb = lbase0 + (unsigned)(((c + 1) & 1) * CHUNK_BYTES);
            const unsigned gb = (unsigned)(c + 1) * (unsigned)CHUNK_BYTES;
            #pragma unroll
            for (int i = 0; i < 16; ++i) {
                unsigned u = (unsigned)tid + 256u * (unsigned)i;
                async_ld16(lb + u * 16u, gb + u * 16u, wbase);
            }
            wait_async16();     // the 16 older loads (chunk c) are complete
        } else {
            wait_async0();
        }
        __syncthreads();        // chunk c visible to all waves

        const unsigned short* buf = sbuf + (size_t)(c & 1) * CHUNK_SH;
        #pragma unroll
        for (int tt = 0; tt < 4; ++tt) {
            const int nl = ch * 64 + tt * 16;   // local code index base in chunk
            // B fragment from LDS (same k order as A: lanes 0-15 K 0..7,16..23; 16-31 K 8..15,24..31)
            const unsigned short* brow = buf + (size_t)(nl + lrow) * D_ + hi * 8;
            BFrag Bf[8];
            #pragma unroll
            for (int cc = 0; cc < 8; ++cc) {
                Bf[cc].q[0] = *reinterpret_cast<const U4*>(brow + cc * 32);
                Bf[cc].q[1] = *reinterpret_cast<const U4*>(brow + cc * 32 + 16);
            }
            v8f acc = {};
            #pragma unroll
            for (int cc = 0; cc < 8; ++cc)
                acc = __builtin_amdgcn_wmma_f32_16x16x32_bf16(
                    false, A[cc].v, false, Bf[cc].v, (short)0, acc, false, false);

            const int col = c * CHUNK_CODES + nl + lrow;
            const float wn = w2[col];
            #pragma unroll
            for (int r = 0; r < 8; ++r) {
                float dd = wn - 2.0f * acc[r];     // ||f||^2 added at the end
                if (dd < minv[r] || (dd == minv[r] && col < mini[r])) { minv[r] = dd; mini[r] = col; }
            }
        }
        __syncthreads();        // done reading buf[c&1] before it is refilled
    }

    // reduce argmin across the 16 lanes of each half-wave
    #pragma unroll
    for (int r = 0; r < 8; ++r) {
        float v = minv[r]; int i = mini[r];
        #pragma unroll
        for (int m = 8; m >= 1; m >>= 1) {
            float ov = __shfl_xor(v, m, 32);
            int   oi = __shfl_xor(i, m, 32);
            if (ov < v || (ov == v && oi < i)) { v = ov; i = oi; }
        }
        if (lrow == 0) {
            const int row = r + hi * 8;           // C/D layout: lanes 16-31 hold M=8..15
            sv[wave * 16 + row] = v;
            si[wave * 16 + row] = i;
        }
    }
    __syncthreads();
    if (tid < 64) {                                // combine the two column halves
        const int rg2 = tid >> 4, rr = tid & 15;
        float v = sv[rg2 * 16 + rr];       int i = si[rg2 * 16 + rr];
        float ov = sv[(rg2 + 4) * 16 + rr]; int oi = si[(rg2 + 4) * 16 + rr];
        if (ov < v || (ov == v && oi < i)) { v = ov; i = oi; }
        minv_out[m0 + tid] = v + f2[m0 + tid];
        mini_out[m0 + tid] = i;
    }
}

// ---------------- 4: per-batch-row stable bitonic argsort + gather ----------------
__global__ __launch_bounds__(256) void vq_sort_gather_kernel(
    const float* __restrict__ minv, const int* __restrict__ mini,
    int* __restrict__ enc, float* __restrict__ enc_f)
{
    __shared__ float sv[S_];
    __shared__ int   si[S_];
    const int bb = blockIdx.x, tid = threadIdx.x;
    #pragma unroll
    for (int t = 0; t < 4; ++t) {
        int p = tid + 256 * t;
        sv[p] = minv[bb * S_ + p];
        si[p] = p;
    }
    __syncthreads();
    for (int k = 2; k <= S_; k <<= 1) {
        for (int j = k >> 1; j > 0; j >>= 1) {
            #pragma unroll
            for (int t = 0; t < 4; ++t) {
                int i = tid + 256 * t;
                int ixj = i ^ j;
                if (ixj > i) {
                    bool up = ((i & k) == 0);
                    float vi = sv[i], vj = sv[ixj];
                    int   ii = si[i], ij = si[ixj];
                    bool gt = (vi > vj) || (vi == vj && ii > ij);   // stable tie-break
                    if (gt == up) { sv[i] = vj; sv[ixj] = vi; si[i] = ij; si[ixj] = ii; }
                }
            }
            __syncthreads();
        }
    }
    #pragma unroll
    for (int t = 0; t < 4; ++t) {
        int p = tid + 256 * t;
        int ord = si[p];                 // in [0, S)
        int e = mini[ord];               // faithful reference bug: flat gather, batch-0 rows
        enc[bb * S_ + p] = e;
        enc_f[bb * S_ + p] = (float)e;
    }
}

// ---------------- 5: quantize + MSE + counts + dw ----------------
__global__ __launch_bounds__(256) void vq_quant_kernel(
    const float* __restrict__ x, const float* __restrict__ weight,
    const int* __restrict__ enc, float* __restrict__ outq,
    float* __restrict__ dw, float* __restrict__ counts, float* __restrict__ mse)
{
    __shared__ float red[8];
    const int n = blockIdx.x, d = threadIdx.x;
    const int k = enc[n];
    const float xi = x[(size_t)n * D_ + d];
    const float q  = weight[(size_t)k * D_ + d];
    outq[(size_t)n * D_ + d] = q;
    const float diff = q - xi;
    float s = diff * diff;
    #pragma unroll
    for (int m = 16; m >= 1; m >>= 1) s += __shfl_xor(s, m, 32);
    if ((d & 31) == 0) red[d >> 5] = s;
    __syncthreads();
    if (d == 0) {
        float t = 0.f; for (int i = 0; i < 8; ++i) t += red[i];
        atomicAdd(mse, t);
        atomicAdd(&counts[k], 1.0f);
    }
    atomicAdd(&dw[(size_t)k * D_ + d], xi);
}

// ---------------- 6: cluster sizes, loss, perplexity ----------------
__global__ __launch_bounds__(256) void vq_finalize_cs_kernel(
    const float* __restrict__ counts, const float* __restrict__ ema_cs,
    const float* __restrict__ mse, float* __restrict__ out_cs,
    float* __restrict__ out_loss, float* __restrict__ out_perp)
{
    __shared__ float redn[8], rede[8];
    __shared__ float nsum_s;
    const int tid = threadIdx.x;
    float raw[4]; float ns = 0.f, ent = 0.f;
    #pragma unroll
    for (int t = 0; t < 4; ++t) {
        int kk = tid + 256 * t;
        float c = counts[kk];
        raw[t] = ema_cs[kk] * DECAY_ + (1.0f - DECAY_) * c;
        ns += raw[t];
        float p = c / (float)N_;
        ent += p * logf(p + 1e-10f);
    }
    #pragma unroll
    for (int m = 16; m >= 1; m >>= 1) { ns += __shfl_xor(ns, m, 32); ent += __shfl_xor(ent, m, 32); }
    if ((tid & 31) == 0) { redn[tid >> 5] = ns; rede[tid >> 5] = ent; }
    __syncthreads();
    if (tid == 0) {
        float tn = 0.f, te = 0.f;
        for (int i = 0; i < 8; ++i) { tn += redn[i]; te += rede[i]; }
        nsum_s = tn;
        *out_perp = expf(-te);
        *out_loss = (1.0f + COMMIT_) * mse[0] / (float)((size_t)N_ * D_);
    }
    __syncthreads();
    const float nsum = nsum_s;
    #pragma unroll
    for (int t = 0; t < 4; ++t) {
        int kk = tid + 256 * t;
        out_cs[kk] = (raw[t] + EPS_) / (nsum + (float)K_ * EPS_) * nsum;
    }
}

// ---------------- 7: new_ema_w / new_weight ----------------
__global__ void vq_finalize_w_kernel(
    const float* __restrict__ ema_w, const float* __restrict__ dw,
    const float* __restrict__ new_cs, float* __restrict__ out_w, float* __restrict__ out_ema)
{
    size_t i = (size_t)blockIdx.x * blockDim.x + threadIdx.x;
    if (i >= (size_t)K_ * D_) return;
    float e = ema_w[i] * DECAY_ + (1.0f - DECAY_) * dw[i];
    out_ema[i] = e;
    out_w[i] = e / new_cs[i / D_];
}

// ---------------- launch ----------------
extern "C" void kernel_launch(void* const* d_in, const int* in_sizes, int n_in,
                              void* d_out, int out_size, void* d_ws, size_t ws_size,
                              hipStream_t stream) {
    const float* x      = (const float*)d_in[0];   // [B,S,D]
    const float* weight = (const float*)d_in[1];   // [K,D]
    const float* ema_w  = (const float*)d_in[2];   // [K,D]
    const float* ema_cs = (const float*)d_in[3];   // [K]
    const float* gamma  = (const float*)d_in[4];   // [D]
    const float* beta   = (const float*)d_in[5];   // [D]

    char* ws = (char*)d_ws;
    const size_t OFF_FEATS  = 0;                                    // N*D bf16
    const size_t OFF_WBF    = OFF_FEATS + (size_t)N_ * D_ * 2;      // K*D bf16
    const size_t OFF_F2     = OFF_WBF   + (size_t)K_ * D_ * 2;      // N f32
    const size_t OFF_W2     = OFF_F2    + (size_t)N_ * 4;           // K f32
    const size_t OFF_MINV   = OFF_W2    + (size_t)K_ * 4;           // N f32
    const size_t OFF_MINI   = OFF_MINV  + (size_t)N_ * 4;           // N i32
    const size_t OFF_ENC    = OFF_MINI  + (size_t)N_ * 4;           // N i32
    const size_t OFF_COUNTS = OFF_ENC   + (size_t)N_ * 4;           // K f32   (zeroed)
    const size_t OFF_DW     = OFF_COUNTS+ (size_t)K_ * 4;           // K*D f32 (zeroed)
    const size_t OFF_MSE    = OFF_DW    + (size_t)K_ * D_ * 4;      // 1 f32   (zeroed)

    unsigned short* featsBF = (unsigned short*)(ws + OFF_FEATS);
    unsigned short* wBF     = (unsigned short*)(ws + OFF_WBF);
    float* f2     = (float*)(ws + OFF_F2);
    float* w2     = (float*)(ws + OFF_W2);
    float* minv   = (float*)(ws + OFF_MINV);
    int*   mini   = (int*)  (ws + OFF_MINI);
    int*   enc    = (int*)  (ws + OFF_ENC);
    float* counts = (float*)(ws + OFF_COUNTS);
    float* dw     = (float*)(ws + OFF_DW);
    float* mse    = (float*)(ws + OFF_MSE);

    float* out = (float*)d_out;
    float* out_q    = out;                                   // N*D
    float* out_loss = out + (size_t)N_ * D_;                 // 1
    float* out_perp = out_loss + 1;                          // 1
    float* out_enc  = out_perp + 1;                          // N
    float* out_w    = out_enc + N_;                          // K*D
    float* out_ema  = out_w + (size_t)K_ * D_;               // K*D
    float* out_cs   = out_ema + (size_t)K_ * D_;             // K

    {
        int nz = K_ + K_ * D_ + 1;
        vq_zero_kernel<<<(nz + 255) / 256, 256, 0, stream>>>(counts, nz);
    }
    vq_ln_kernel<<<N_, 256, 0, stream>>>(x, gamma, beta, featsBF, f2);
    vq_wprep_kernel<<<K_, 256, 0, stream>>>(weight, wBF, w2);
    vq_dist_argmin_kernel<<<N_ / 64, 256, 2 * CHUNK_BYTES, stream>>>(featsBF, f2, wBF, w2, minv, mini);
    vq_sort_gather_kernel<<<B_, 256, 0, stream>>>(minv, mini, enc, out_enc);
    vq_quant_kernel<<<N_, 256, 0, stream>>>(x, weight, enc, out_q, dw, counts, mse);
    vq_finalize_cs_kernel<<<1, 256, 0, stream>>>(counts, ema_cs, mse, out_cs, out_loss, out_perp);
    {
        int ntot = K_ * D_;
        vq_finalize_w_kernel<<<(ntot + 255) / 256, 256, 0, stream>>>(ema_w, dw, out_cs, out_w, out_ema);
    }
}
